// LapDecoder_39599598469812
// MI455X (gfx1250) — matrix operand order repaired and checked by reading, not compiled, not run on hardware
//
#include <hip/hip_runtime.h>
#include <stdint.h>

typedef _Float16 f16;
typedef __attribute__((ext_vector_type(16))) _Float16 v16h;
typedef __attribute__((ext_vector_type(8)))  float    v8f;
typedef __attribute__((ext_vector_type(4)))  uint32_t v4u;
typedef __attribute__((ext_vector_type(8)))  uint32_t v8u;

#define B_  4
#define N_  50000
#define E_  800000
#define C_  128
#define LDS_ROW 136  // 128 f16 + 8 f16 TDM pad -> 272B stride (16B aligned, bank-spread)

// ---------------------------------------------------------------- input proj
__global__ void in_proj(const float* __restrict__ inp, const float* __restrict__ noi,
                        const float* __restrict__ Wi, const float* __restrict__ bi,
                        const float* __restrict__ Wn, const float* __restrict__ bn,
                        float* __restrict__ out) {
  __shared__ float sIn[3];
  __shared__ float sNo[100];
  long row = blockIdx.x;
  if (threadIdx.x < 3)   sIn[threadIdx.x] = inp[row * 3 + threadIdx.x];
  if (threadIdx.x < 100) sNo[threadIdx.x] = noi[row * 100 + threadIdx.x];
  __syncthreads();
  int c = threadIdx.x;  // 128 threads
  float acc = bi[c] + bn[c];
#pragma unroll
  for (int k = 0; k < 3; k++)   acc += sIn[k] * Wi[k * C_ + c];
  for (int k = 0; k < 100; k++) acc += sNo[k] * Wn[k * C_ + c];
  out[row * C_ + c] = acc;
}

// ---------------------------------------------------------------- elu + pack
__global__ void elu_pack(const float* __restrict__ x, f16* __restrict__ e16,
                         f16* __restrict__ lzero, float* __restrict__ stats,
                         long total, int zeroL) {
  long stride = (long)gridDim.x * blockDim.x;
  for (long i = (long)blockIdx.x * blockDim.x + threadIdx.x; i < total; i += stride) {
    float v = x[i];
    float e = v > 0.f ? v : (__expf(v) - 1.0f);
    e16[i] = (f16)e;
    if (zeroL) lzero[i] = (f16)0.0f;
  }
  if (blockIdx.x == 0 && threadIdx.x < 512) stats[threadIdx.x] = 0.0f;
}

// ---------------------------------------------------------------- L @ x (COO)
__global__ void lap_scatter(const int* __restrict__ rows, const int* __restrict__ cols,
                            const float* __restrict__ vals, const f16* __restrict__ e16,
                            f16* __restrict__ L16) {
  long tid = (long)blockIdx.x * blockDim.x + threadIdx.x;
  long ei  = tid >> 5;
  if (ei >= (long)B_ * E_) return;
  int lane = threadIdx.x & 31;
  long b = ei / E_;
  float val = vals[ei];
  long col = cols[ei];
  long row = rows[ei];
  long src = (b * N_ + col) * C_ + lane * 4;
  long dstbyte = ((b * N_ + row) * C_ + lane * 4) * 2;

  union { uint32_t u; _Float16 h[2]; } a0, a1, p0, p1;
  const uint32_t* sp = (const uint32_t*)(e16 + src);
  a0.u = sp[0];
  a1.u = sp[1];
  p0.h[0] = (_Float16)((float)a0.h[0] * val);
  p0.h[1] = (_Float16)((float)a0.h[1] * val);
  p1.h[0] = (_Float16)((float)a1.h[0] * val);
  p1.h[1] = (_Float16)((float)a1.h[1] * val);

  uint64_t base = (uint64_t)(uintptr_t)L16;
  asm volatile("global_atomic_pk_add_f16 %0, %1, %2"
               :: "v"((uint32_t)dstbyte), "v"(p0.u), "s"(base) : "memory");
  asm volatile("global_atomic_pk_add_f16 %0, %1, %2"
               :: "v"((uint32_t)(dstbyte + 4)), "v"(p1.u), "s"(base) : "memory");
}

// ---------------------------------------------------------------- BN stats
#define STAT_ROWS 512
__global__ void bn_stats(const f16* __restrict__ A0, const f16* __restrict__ A1,
                         float* __restrict__ stats, long Mtot) {
  int c = threadIdx.x;  // 256 threads -> 256 channels
  const f16* src = (c < C_) ? (A0 + c) : (A1 + (c - C_));
  long r0 = (long)blockIdx.x * STAT_ROWS;
  long r1 = r0 + STAT_ROWS; if (r1 > Mtot) r1 = Mtot;
  float s = 0.f, s2 = 0.f;
  for (long r = r0; r < r1; r++) {
    float v = (float)src[r * C_];
    s += v; s2 += v * v;
  }
  atomicAdd(&stats[c], s);
  atomicAdd(&stats[256 + c], s2);
}

// ---------------------------------------------------------------- BN fold
__global__ void bn_fold(const float* __restrict__ stats, const float* __restrict__ gamma,
                        const float* __restrict__ beta, const float* __restrict__ W,
                        const float* __restrict__ b, f16* __restrict__ Wf,
                        float* __restrict__ bf, long Mtot, int Kch) {
  int n = threadIdx.x;  // 128 threads
  float inv = 1.0f / (float)Mtot;
  float acc = 0.f;
  for (int k = 0; k < Kch; k++) {
    float mean = stats[k] * inv;
    float var  = stats[256 + k] * inv - mean * mean;
    float a = gamma[k] * rsqrtf(var + 1e-5f);
    float t = beta[k] - mean * a;
    float w = W[k * C_ + n];
    Wf[k * C_ + n] = (f16)(a * w);
    acc += t * w;
  }
  bf[n] = b[n] + acc;
}

// ---------------------------------------------------------------- WMMA GEMM
union AFrag { v16h v; uint32_t u[8]; };
union BFrag { v16h v; uint4 q[2]; };

__device__ __forceinline__ AFrag load_afrag(const f16* __restrict__ A0,
                                            const f16* __restrict__ A1,
                                            long row, int kh, int kkg) {
  // ISA 16-bit A 16x32 layout: lane half kh supplies K = kh*8..+7 and 16+kh*8..+7
  const f16* src = (kkg < C_) ? (A0 + kkg) : (A1 + (kkg - C_));
  const f16* ap = src + row * C_ + kh * 8;
  AFrag a;
  *(uint4*)&a.u[0] = *(const uint4*)ap;         // K = kh*8 .. +7
  *(uint4*)&a.u[4] = *(const uint4*)(ap + 16);  // K = 16 + kh*8 .. +7
  return a;
}

// out[M x 128] = concat(A0,A1)[M x KTOT](f16) @ Wf[KTOT x 128](f16) + bf (+res)
// Weights staged LDS-side by the Tensor Data Mover (one TDM op per block).
template <int KTOT, bool HAS_RES>
__global__ void gemm_wmma(const f16* __restrict__ A0, const f16* __restrict__ A1,
                          const f16* __restrict__ Wf, const float* __restrict__ bfold,
                          const float* __restrict__ res, float* __restrict__ out,
                          long Mtot) {
  __shared__ f16 ldsW[256 * LDS_ROW];  // padded rows: 272B stride

  if (threadIdx.x < 32) {  // wave 0 issues the TDM descriptor (wave-uniform branch)
    uint64_t ga = (uint64_t)(uintptr_t)Wf;
    uint32_t lbase = (uint32_t)(uintptr_t)ldsW;
    v4u g0;
    g0.x = 1u;                                              // count=1, user mode
    g0.y = lbase;                                           // lds_addr
    g0.z = (uint32_t)ga;                                    // global_addr[95:64]
    g0.w = (uint32_t)((ga >> 32) & 0x1FFFFFFu) | (2u << 30);// addr hi | type=2
    v8u g1;
    g1.s0 = (1u << 16)        // data_size = 2 bytes
          | (1u << 20)        // pad_enable
          | (5u << 22)        // pad_interval: 64 DWORDs (one 256B row)
          | (3u << 25);       // pad_amount: 4 DWORDs (16B) -> 272B LDS row stride
    g1.s1 = (128u & 0xFFFFu) << 16;              // tensor_dim0 = 128 (lo16)
    g1.s2 = ((uint32_t)KTOT & 0xFFFFu) << 16;    // dim0 hi16=0 | tensor_dim1 lo16
    g1.s3 = (128u << 16);                        // dim1 hi16=0 | tile_dim0 = 128
    g1.s4 = (uint32_t)KTOT & 0xFFFFu;            // tile_dim1 = KTOT, tile_dim2 = 0
    g1.s5 = 128u;                                // tensor_dim0_stride lo32
    g1.s6 = 0u;
    g1.s7 = 0u;
    asm volatile("tensor_load_to_lds %0, %1" :: "s"(g0), "s"(g1) : "memory");
    __builtin_amdgcn_s_wait_tensorcnt(0);
  }
  __syncthreads();

  int wave = threadIdx.x >> 5;
  int lane = threadIdx.x & 31;
  long rowbase = (long)blockIdx.x * 128 + wave * 16;
  int m  = lane & 15;
  int kh = lane >> 4;
  long arow = rowbase + m;
  if (arow >= Mtot) arow = Mtot - 1;  // clamp: keeps EXEC all-ones for WMMA

  v8f acc[8] = {};

#pragma unroll
  for (int kk = 0; kk < KTOT; kk += 32) {               // fully unrolled (static)
    AFrag a = load_afrag(A0, A1, arow, kh, kk);
    const f16* bp = ldsW + (long)(kk + kh * 8) * LDS_ROW + (lane & 15); // lane%16 = N col, K = kh*8.. / 16+kh*8..
    BFrag bt[8];
#pragma unroll
    for (int t = 0; t < 8; t++) {                       // batch all LDS loads,
#pragma unroll
      for (int s = 0; s < 8; s++) {                     // one wait for the group
        bt[t].v[s]     = bp[s * LDS_ROW + t * 16];
        bt[t].v[8 + s] = bp[(16 + s) * LDS_ROW + t * 16];
      }
    }
#pragma unroll
    for (int t = 0; t < 8; t++) {                       // back-to-back WMMAs
      acc[t] = __builtin_amdgcn_wmma_f32_16x16x32_f16(
          false, a.v, false, bt[t].v, (short)0, acc[t], false, false);
    }
  }

#pragma unroll
  for (int t = 0; t < 8; t++) {
    int n = t * 16 + (lane & 15);
    float bias = bfold[n];
#pragma unroll
    for (int r = 0; r < 8; r++) {
      long row = rowbase + r + 8 * kh;  // C/D layout: vgpr r -> M=r (+8 hi lanes)
      if (row < Mtot) {
        long idx = row * C_ + n;
        float v = acc[t][r] + bias;
        if (HAS_RES) v += res[idx];
        out[idx] = v;
      }
    }
  }
}

// ---------------------------------------------------------------- head
__global__ void mu_out(const f16* __restrict__ e16, const float* __restrict__ Wmu,
                       const float* __restrict__ bmu, const float* __restrict__ inp,
                       const float* __restrict__ logvar, float* __restrict__ out, long Mtot) {
  long row = (long)blockIdx.x * blockDim.x + threadIdx.x;
  if (row >= Mtot) return;
  float a0 = bmu[0], a1 = bmu[1], a2 = bmu[2];
  const f16* xr = e16 + row * C_;
#pragma unroll 4
  for (int k = 0; k < C_; k++) {
    float v = (float)xr[k];
    a0 += v * Wmu[k * 3 + 0];
    a1 += v * Wmu[k * 3 + 1];
    a2 += v * Wmu[k * 3 + 2];
  }
  out[row * 3 + 0] = a0 + inp[row * 3 + 0];
  out[row * 3 + 1] = a1 + inp[row * 3 + 1];
  out[row * 3 + 2] = a2 + inp[row * 3 + 2];
  float lv = logvar[0];
  long o = Mtot * 3 + row * 3;
  out[o + 0] = lv; out[o + 1] = lv; out[o + 2] = lv;
}

// ---------------------------------------------------------------- launcher
extern "C" void kernel_launch(void* const* d_in, const int* in_sizes, int n_in,
                              void* d_out, int out_size, void* d_ws, size_t ws_size,
                              hipStream_t stream) {
  const float* inputs   = (const float*)d_in[0];
  const float* noise    = (const float*)d_in[1];
  const int*   L_rows   = (const int*)d_in[3];
  const int*   L_cols   = (const int*)d_in[4];
  const float* L_vals   = (const float*)d_in[5];
  const float* W_in     = (const float*)d_in[6];
  const float* b_in     = (const float*)d_in[7];
  const float* W_noise  = (const float*)d_in[8];
  const float* b_noise  = (const float*)d_in[9];
  const float* rn_gamma = (const float*)d_in[10];
  const float* rn_beta  = (const float*)d_in[11];
  const float* rn_W     = (const float*)d_in[12];
  const float* rn_b     = (const float*)d_in[13];
  const float* bn2_g    = (const float*)d_in[14];
  const float* bn2_b    = (const float*)d_in[15];
  const float* W2       = (const float*)d_in[16];
  const float* b2       = (const float*)d_in[17];
  const float* W_mu     = (const float*)d_in[18];
  const float* b_mu     = (const float*)d_in[19];
  const float* logvar   = (const float*)d_in[20];
  (void)in_sizes; (void)n_in; (void)out_size; (void)ws_size;

  const long BN = (long)B_ * N_;
  char* ws = (char*)d_ws;
  size_t off = 0;
  auto take = [&](size_t bytes) { char* p = ws + off; off = (off + bytes + 255) & ~(size_t)255; return p; };
  float* R    = (float*)take(BN * C_ * 4);
  float* T    = (float*)take(BN * C_ * 4);
  f16*   E16  = (f16*)  take(BN * C_ * 2);
  f16*   L16  = (f16*)  take(BN * C_ * 2);
  f16*   Wf   = (f16*)  take(2 * C_ * C_ * 2);
  float* bf   = (float*)take(C_ * 4);
  float* stats= (float*)take(512 * 4);

  const long total = BN * C_;
  const int eluBlocks  = (int)((total + 255) / 256);
  const int edgeBlocks = (int)(((long)B_ * E_ * 32 + 255) / 256);
  const int statBlocks = (int)((BN + STAT_ROWS - 1) / STAT_ROWS);
  const unsigned gemmBlocks = (unsigned)((BN + 127) / 128);

  in_proj<<<(unsigned)BN, 128, 0, stream>>>(inputs, noise, W_in, b_in, W_noise, b_noise, R);

  for (int i = 0; i < 5; i++) {
    for (int j = 0; j < 2; j++) {
      int l = i * 2 + j;
      const float* xin = (j == 0) ? R : T;
      elu_pack<<<eluBlocks, 256, 0, stream>>>(xin, E16, L16, stats, total, 1);
      lap_scatter<<<edgeBlocks, 256, 0, stream>>>(L_rows, L_cols, L_vals, E16, L16);
      bn_stats<<<statBlocks, 256, 0, stream>>>(E16, L16, stats, BN);
      bn_fold<<<1, 128, 0, stream>>>(stats, rn_gamma + l * 256, rn_beta + l * 256,
                                     rn_W + (size_t)l * 256 * C_, rn_b + l * C_,
                                     Wf, bf, BN, 256);
      if (j == 0)
        gemm_wmma<256, false><<<gemmBlocks, 256, 0, stream>>>(E16, L16, Wf, bf, nullptr, T, BN);
      else
        gemm_wmma<256, true><<<gemmBlocks, 256, 0, stream>>>(E16, L16, Wf, bf, R, R, BN);
    }
  }

  // head: elu -> BN(fold) -> W2 -> elu -> W_mu (+inputs); logvar broadcast
  elu_pack<<<eluBlocks, 256, 0, stream>>>(R, E16, L16, stats, total, 0);
  bn_stats<<<statBlocks, 256, 0, stream>>>(E16, E16, stats, BN);
  bn_fold<<<1, 128, 0, stream>>>(stats, bn2_g, bn2_b, W2, b2, Wf, bf, BN, 128);
  gemm_wmma<128, false><<<gemmBlocks, 256, 0, stream>>>(E16, E16, Wf, bf, nullptr, T, BN);
  elu_pack<<<eluBlocks, 256, 0, stream>>>(T, E16, L16, stats, total, 0);
  mu_out<<<(unsigned)((BN + 255) / 256), 256, 0, stream>>>(E16, W_mu, b_mu, inputs,
                                                           logvar, (float*)d_out, BN);
}
